// HungarianMatcher_41961830481930
// MI455X (gfx1250) — matrix operand order repaired and checked by reading, not compiled, not run on hardware
//
#include <hip/hip_runtime.h>

// ---------------------------------------------------------------------------
// HungarianMatcher similarity stage for MI455X (gfx1250, wave32, WMMA).
// B=2, N=128, K=64, HW=65536, C=134.
//  inter[b,n,k] = sum_p A[b,n,p]*T[b,k,p]   (GEMM, ~2.15 GFLOP, ~100 MB traffic)
//  sim = classProb[b,n,tc[b,k]] * (2*inter+eps)/(rowsumA+rowsumT+eps)
// HBM-bound at 23.3 TB/s (~4.3us floor) if matrix math runs on bf16 WMMA.
// target_mask is binary -> exact in bf16; input_mask bf16 rounding is random
// sign, ~1e-5 relative after 64K-term f32 accumulation. Row sums use f32.
// ---------------------------------------------------------------------------

typedef __attribute__((ext_vector_type(16))) __bf16 v16bf;
typedef __attribute__((ext_vector_type(8)))  float  v8f;

#define B_   2
#define N_   128
#define K_   64
#define HW_  65536
#define C_   134
#define CH   64      // HW columns staged per LDS chunk
#define CHP  72      // padded LDS row stride (bf16 units) -> conflict-free frag reads
#define EPSV 1e-5f

// round-to-nearest-even f32 -> bf16
static __device__ __forceinline__ unsigned short f2bf(float f) {
    unsigned int u = __float_as_uint(f);
    u += 0x7FFFu + ((u >> 16) & 1u);
    return (unsigned short)(u >> 16);
}

// Stage 1: one workgroup per (batch, HW-slice). 8 waves; wave w computes
// output rows [16w,16w+16) x all 64 cols via v_wmma_f32_16x16x32_bf16.
__global__ __launch_bounds__(256)
void hm_gemm_stage(const float* __restrict__ Amask,
                   const float* __restrict__ Tmask,
                   float* __restrict__ wsC,   // [B*S][128*64] partial intersections
                   float* __restrict__ wsSA,  // [B*S][128]    partial row sums of A
                   float* __restrict__ wsST,  // [B*S][64]     partial row sums of T
                   int S)
{
    __shared__ unsigned short As[N_ * CHP];   // bf16 A chunk (128 x 64, pad 8)
    __shared__ unsigned short Ts[K_ * CHP];   // bf16 T chunk ( 64 x 64, pad 8)
    __shared__ float pA[N_ * 16];             // per-thread row-sum partials
    __shared__ float pT[K_ * 16];

    const int tid  = threadIdx.x;
    const int wave = tid >> 5;
    const int lane = tid & 31;
    const int b    = blockIdx.x / S;
    const int s    = blockIdx.x % S;
    const int len  = HW_ / S;                 // slice length (multiple of CH)
    const int cbeg = s * len;

    const float* Ab = Amask + (size_t)b * N_ * HW_;
    const float* Tb = Tmask + (size_t)b * K_ * HW_;

    float sa[8], st[4];
#pragma unroll
    for (int i = 0; i < 8; ++i) sa[i] = 0.f;
#pragma unroll
    for (int i = 0; i < 4; ++i) st[i] = 0.f;

    const v8f vzero = {0.f, 0.f, 0.f, 0.f, 0.f, 0.f, 0.f, 0.f};
    v8f acc[4];
#pragma unroll
    for (int kt = 0; kt < 4; ++kt) acc[kt] = vzero;

    const int m  = lane & 15;   // row-within-tile / output column
    const int hh = lane >> 4;   // lane half selector

    for (int c0 = cbeg; c0 < cbeg + len; c0 += CH) {
        __syncthreads();  // protect LDS reuse from previous iteration's reads

        // Cooperative load A chunk: 128 rows x 16 float4 segs = 2048 float4.
#pragma unroll
        for (int i = 0; i < 8; ++i) {
            const int t = tid + i * 256;
            const int row = t >> 4, seg = t & 15;
            const float4 v = *(const float4*)(Ab + (size_t)row * HW_ + c0 + seg * 4);
            unsigned short* d = &As[row * CHP + seg * 4];
            d[0] = f2bf(v.x); d[1] = f2bf(v.y); d[2] = f2bf(v.z); d[3] = f2bf(v.w);
            sa[i] += (v.x + v.y) + (v.z + v.w);   // exact-f32 cardinality partials
        }
        // Cooperative load T chunk: 64 rows x 16 float4 segs = 1024 float4.
#pragma unroll
        for (int i = 0; i < 4; ++i) {
            const int t = tid + i * 256;
            const int row = t >> 4, seg = t & 15;
            const float4 v = *(const float4*)(Tb + (size_t)row * HW_ + c0 + seg * 4);
            unsigned short* d = &Ts[row * CHP + seg * 4];
            d[0] = f2bf(v.x); d[1] = f2bf(v.y); d[2] = f2bf(v.z); d[3] = f2bf(v.w);
            st[i] += (v.x + v.y) + (v.z + v.w);
        }
        __syncthreads();

        // 2 K-steps of 32 per chunk; 4 column tiles -> 8 WMMAs / wave / chunk.
#pragma unroll
        for (int ks = 0; ks < CH / 32; ++ks) {
            const int koff = ks * 32;
            // A fragment, 16x32 bf16 (ISA 7.12.2): lane half picks K+0/K+8,
            // VGPRs 0-3 -> K 0..15 region, VGPRs 4-7 -> K 16..31 region.
            union { v16bf v; unsigned int u[8]; } af;
            const unsigned short* arow = &As[(wave * 16 + m) * CHP];
#pragma unroll
            for (int vg = 0; vg < 8; ++vg) {
                const int kk = koff + ((vg & 3) * 2) + ((vg >> 2) * 16) + hh * 8;
                af.u[vg] = *(const unsigned int*)(arow + kk);
            }
#pragma unroll
            for (int kt = 0; kt < 4; ++kt) {
                // B fragment, 32x16 bf16: same K order as A (lane half picks K+0/K+8, VGPRs 4-7 -> K+16),
                // column = lane&15; B[kd][col] = T[kt*16+col][p].
                union { v16bf v; unsigned int u[8]; } bf;
                const unsigned short* trow = &Ts[(kt * 16 + m) * CHP];
#pragma unroll
                for (int vg = 0; vg < 8; ++vg) {
                    const int kk = koff + ((vg & 3) * 2) + ((vg >> 2) * 16) + hh * 8;
                    bf.u[vg] = *(const unsigned int*)(trow + kk);
                }
                acc[kt] = __builtin_amdgcn_wmma_f32_16x16x32_bf16(
                    false, af.v, false, bf.v, (short)0, acc[kt], false, false);
            }
        }
    }

    // Deterministic row-sum reduction through LDS (no atomics).
    __syncthreads();
#pragma unroll
    for (int i = 0; i < 8; ++i) pA[tid + i * 256] = sa[i];  // slot = row*16+seg
#pragma unroll
    for (int i = 0; i < 4; ++i) pT[tid + i * 256] = st[i];
    __syncthreads();

    const size_t slice = (size_t)b * S + s;
    if (tid < N_) {
        float v = 0.f;
#pragma unroll
        for (int j = 0; j < 16; ++j) v += pA[tid * 16 + j];
        wsSA[slice * N_ + tid] = v;
    }
    if (tid < K_) {
        float v = 0.f;
#pragma unroll
        for (int j = 0; j < 16; ++j) v += pT[tid * 16 + j];
        wsST[slice * K_ + tid] = v;
    }

    // Write partial C tile. D layout: VGPR r, lane half -> M = 16w + hh*8 + r,
    // column = kt*16 + (lane&15).
    float* cdst = wsC + slice * (size_t)(N_ * K_);
#pragma unroll
    for (int kt = 0; kt < 4; ++kt) {
#pragma unroll
        for (int r = 0; r < 8; ++r) {
            const int row = wave * 16 + hh * 8 + r;
            const int col = kt * 16 + m;
            cdst[row * K_ + col] = acc[kt][r];
        }
    }
}

// Stage 2: reduce split-HW partials, gather class prob, emit dice*class sim.
__global__ __launch_bounds__(256)
void hm_combine(const float* __restrict__ wsC,
                const float* __restrict__ wsSA,
                const float* __restrict__ wsST,
                const float* __restrict__ classProb,
                const int*  __restrict__ targetClass,
                float* __restrict__ out,
                int S)
{
    const int idx = blockIdx.x * 256 + threadIdx.x;
    if (idx >= B_ * N_ * K_) return;
    const int b = idx / (N_ * K_);
    const int r = idx - b * (N_ * K_);
    const int n = r / K_;
    const int k = r - n * K_;

    float inter = 0.f, suma = 0.f, sumt = 0.f;
    for (int s = 0; s < S; ++s) {
        const size_t sl = (size_t)b * S + s;
        inter += wsC[sl * (size_t)(N_ * K_) + (size_t)n * K_ + k];
        suma  += wsSA[sl * N_ + n];
        sumt  += wsST[sl * K_ + k];
    }
    const int tc = targetClass[b * K_ + k];
    const float sc = classProb[((size_t)b * N_ + n) * C_ + tc];
    out[idx] = sc * (2.f * inter + EPSV) / ((suma + sumt) + EPSV);
}

extern "C" void kernel_launch(void* const* d_in, const int* in_sizes, int n_in,
                              void* d_out, int out_size, void* d_ws, size_t ws_size,
                              hipStream_t stream) {
    (void)in_sizes; (void)n_in; (void)out_size;
    const float* classProb   = (const float*)d_in[0];  // (B,N,C) f32
    const float* Amask       = (const float*)d_in[1];  // (B,N,HW) f32
    const float* Tmask       = (const float*)d_in[2];  // (B,K,HW) f32
    const int*   targetClass = (const int*)  d_in[3];  // (B,K) int
    // d_in[4] (target_sizes) is unused by the reference similarity math.

    // Split-HW factor: as much parallelism as the workspace allows (det. in ws_size).
    int S = 128;
    const size_t perSlicePair = (size_t)B_ * (N_ * K_ + N_ + K_) * sizeof(float);
    while (S > 1 && (size_t)S * perSlicePair > ws_size) S >>= 1;

    float* wsC  = (float*)d_ws;
    float* wsSA = wsC  + (size_t)B_ * S * N_ * K_;
    float* wsST = wsSA + (size_t)B_ * S * N_;

    hm_gemm_stage<<<B_ * S, 256, 0, stream>>>(Amask, Tmask, wsC, wsSA, wsST, S);

    const int total = B_ * N_ * K_;
    hm_combine<<<(total + 255) / 256, 256, 0, stream>>>(
        wsC, wsSA, wsST, classProb, targetClass, (float*)d_out, S);
}
